// AttentionEstimator_48095043780672
// MI455X (gfx1250) — matrix operand order repaired and checked by reading, not compiled, not run on hardware
//
#include <hip/hip_runtime.h>
#include <hip/hip_bf16.h>

// Fused sharpened-attention for MI455X (gfx1250, wave32, WMMA).
// Identity used: softmax(s)^(1/T) renormalized == softmax(s/T), T=0.5 -> scale = 2/sqrt(512).
// Pass 1: convert Q,K -> bf16 and V -> transposed bf16 (Vt[col][key]) in d_ws (24 MB).
// Pass 2: flash-attention with v_wmma_f32_16x16x32_bf16, online softmax, fp32 accum.

typedef __attribute__((ext_vector_type(16))) __bf16 v16bf;
typedef __attribute__((ext_vector_type(8)))  float  v8f;

#define NKEYS 8192
#define DKDIM 512
#define DVDIM 512
#define BM    32      // query rows per workgroup
#define BN    128     // keys per iteration (16 per wave x 8 waves)
#define SROW  132     // padded row stride of S tile (floats)
#define PROW  136     // padded row stride of P tile (bf16), multiple of 8 for 16B-aligned frags

__device__ __forceinline__ unsigned short f2bf(float f) {
  unsigned int u = __float_as_uint(f);
  u += 0x7FFFu + ((u >> 16) & 1u);        // round-to-nearest-even
  return (unsigned short)(u >> 16);
}

union Frag32B { v16bf v; uint4 u[2]; };

__device__ __forceinline__ v8f wmma_bf16(v16bf a, v16bf b, v8f c) {
  // (neg_a, A, neg_b, B, c_mod, C, reuse_a, reuse_b)
  return __builtin_amdgcn_wmma_f32_16x16x32_bf16(false, a, false, b, (short)0, c, false, false);
}

//--------------------------------------------------------------------------------------
// Convert Q,K (8192x512 f32) -> bf16, flat. 4096 blocks x 256 threads x 4 elems.
//--------------------------------------------------------------------------------------
__global__ __launch_bounds__(256) void cvt_qk_bf16(
    const float* __restrict__ Q, const float* __restrict__ K,
    unsigned short* __restrict__ Qb, unsigned short* __restrict__ Kb)
{
  const size_t i = ((size_t)blockIdx.x * 256 + threadIdx.x) * 4;
  float4 q = *(const float4*)(Q + i);
  float4 k = *(const float4*)(K + i);
  uint2 qo, ko;
  qo.x = (unsigned)f2bf(q.x) | ((unsigned)f2bf(q.y) << 16);
  qo.y = (unsigned)f2bf(q.z) | ((unsigned)f2bf(q.w) << 16);
  ko.x = (unsigned)f2bf(k.x) | ((unsigned)f2bf(k.y) << 16);
  ko.y = (unsigned)f2bf(k.z) | ((unsigned)f2bf(k.w) << 16);
  *(uint2*)(Qb + i) = qo;
  *(uint2*)(Kb + i) = ko;
}

//--------------------------------------------------------------------------------------
// V (8192x512 f32) -> Vt (512x8192 bf16), tiled 32x32 transpose through LDS.
//--------------------------------------------------------------------------------------
__global__ __launch_bounds__(256) void cvt_v_transpose(
    const float* __restrict__ V, unsigned short* __restrict__ Vt)
{
  __shared__ unsigned short tile[32][33];
  const int bx = blockIdx.x & 15;     // 512/32 column tiles
  const int by = blockIdx.x >> 4;     // 8192/32 row tiles
  const int tx = threadIdx.x;         // 0..31
  const int ty = threadIdx.y;         // 0..7
  #pragma unroll
  for (int j = ty; j < 32; j += 8)
    tile[j][tx] = f2bf(V[(size_t)(by * 32 + j) * DVDIM + bx * 32 + tx]);
  __syncthreads();
  #pragma unroll
  for (int j = ty; j < 32; j += 8)
    Vt[(size_t)(bx * 32 + j) * NKEYS + by * 32 + tx] = tile[tx][j];
}

//--------------------------------------------------------------------------------------
// Fused attention. 256 blocks x 256 threads (8 waves). Wave w owns output cols [64w,64w+64).
//--------------------------------------------------------------------------------------
__global__ __launch_bounds__(256) void attn_fused(
    const unsigned short* __restrict__ Qb,
    const unsigned short* __restrict__ Kb,
    const unsigned short* __restrict__ Vt,
    float* __restrict__ out)
{
  __shared__ __align__(16) unsigned short lds_q[BM * DKDIM];   // 32 KB bf16 Q tile
  __shared__ __align__(16) float          lds_s[BM * SROW];    // score tile
  __shared__ __align__(16) unsigned short lds_p[BM * PROW];    // exp tile (bf16)
  __shared__ float lds_m[BM], lds_l[BM], lds_alpha[BM];
  __shared__ float lds_pmax[BM * 8], lds_psum[BM * 8];

  const int tid   = threadIdx.x;
  const int wave  = tid >> 5;
  const int lane  = tid & 31;
  const int l16   = lane & 15;
  const int lh    = lane >> 4;              // lane half: 0 | 1
  const int qrow0 = blockIdx.x * BM;

  { // stage the Q tile: 32x512 bf16 = 2048 uint4, 8 per thread
    const uint4* src = (const uint4*)(Qb + (size_t)qrow0 * DKDIM);
    uint4* dst = (uint4*)lds_q;
    #pragma unroll
    for (int i = 0; i < 8; ++i) dst[tid + 256 * i] = src[tid + 256 * i];
  }
  if (tid < BM) { lds_m[tid] = -3.0e38f; lds_l[tid] = 0.0f; }
  __syncthreads();

  v8f O[2][4] = {};                          // [mfrag][ntile] fp32 accumulators

  // (1/T)/sqrt(DK) * log2(e): softmax done in the exp2 domain (maps to v_exp_f32)
  const float SCALE = (2.0f / 22.62741699796952f) * 1.4426950408889634f;

  for (int nt0 = 0; nt0 < NKEYS; nt0 += BN) {
    //---------------- Phase 1: S = Q K^T for this wave's 16-key slice -----------------
    v8f Sacc[2] = {};
    const unsigned short* krow = Kb + (size_t)(nt0 + wave * 16 + l16) * DKDIM;
    if (nt0 + BN < NKEYS)
      __builtin_prefetch(krow + (size_t)BN * DKDIM, 0, 1);   // global_prefetch_b8
    for (int dk0 = 0; dk0 < DKDIM; dk0 += 32) {
      // B frag (K^T 32x16): lane = key col l16; v0-3: K=lh*8+0..7, v4-7: K=16+lh*8+0..7
      Frag32B B;
      const uint4* kp = (const uint4*)(krow + dk0 + lh * 8);
      B.u[0] = kp[0]; B.u[1] = kp[2];
      #pragma unroll
      for (int mf = 0; mf < 2; ++mf) {
        // A frag (Q 16x32): row = mf*16+l16; v0-3: K=lh*8+0..7, v4-7: K=16+lh*8+0..7
        Frag32B A;
        const uint4* qp = (const uint4*)(lds_q + (mf * 16 + l16) * DKDIM + dk0 + lh * 8);
        A.u[0] = qp[0];
        A.u[1] = qp[2];                     // +16 bf16 elements
        Sacc[mf] = wmma_bf16(A.v, B.v, Sacc[mf]);
      }
    }
    // C layout: lane<16 -> M=r, lane>=16 -> M=8+r; N = l16
    #pragma unroll
    for (int mf = 0; mf < 2; ++mf)
      #pragma unroll
      for (int r = 0; r < 8; ++r)
        lds_s[(mf * 16 + lh * 8 + r) * SROW + wave * 16 + l16] = Sacc[mf][r] * SCALE;
    __syncthreads();

    //---------------- Phase 2: online softmax over the 32x128 tile --------------------
    const int row = tid & 31, seg = tid >> 5;     // 32 rows x 8 segments of 16 cols
    {
      const float* sp = lds_s + row * SROW + seg * 16;
      float mx = sp[0];
      #pragma unroll
      for (int j = 1; j < 16; ++j) mx = fmaxf(mx, sp[j]);
      lds_pmax[row * 8 + seg] = mx;
    }
    __syncthreads();
    if (tid < BM) {
      float mt = lds_pmax[tid * 8];
      #pragma unroll
      for (int j = 1; j < 8; ++j) mt = fmaxf(mt, lds_pmax[tid * 8 + j]);
      float mold = lds_m[tid];
      float mnew = fmaxf(mold, mt);
      lds_m[tid]     = mnew;
      lds_alpha[tid] = __builtin_amdgcn_exp2f(mold - mnew);
    }
    __syncthreads();
    {
      const float mnew = lds_m[row];
      const float* sp = lds_s + row * SROW + seg * 16;
      unsigned short* pp = lds_p + row * PROW + seg * 16;
      float sum = 0.0f;
      #pragma unroll
      for (int j = 0; j < 16; ++j) {
        float p = __builtin_amdgcn_exp2f(sp[j] - mnew);
        sum += p;
        pp[j] = f2bf(p);
      }
      lds_psum[row * 8 + seg] = sum;
    }
    __syncthreads();
    if (tid < BM) {
      float s = 0.0f;
      #pragma unroll
      for (int j = 0; j < 8; ++j) s += lds_psum[tid * 8 + j];
      lds_l[tid] = lds_l[tid] * lds_alpha[tid] + s;
    }
    __syncthreads();

    //---------------- Phase 3: rescale O, accumulate P.V ------------------------------
    float al[2][8];
    #pragma unroll
    for (int mf = 0; mf < 2; ++mf)
      #pragma unroll
      for (int r = 0; r < 8; ++r)
        al[mf][r] = lds_alpha[mf * 16 + lh * 8 + r];
    #pragma unroll
    for (int mf = 0; mf < 2; ++mf)
      #pragma unroll
      for (int nt = 0; nt < 4; ++nt)
        #pragma unroll
        for (int r = 0; r < 8; ++r)
          O[mf][nt][r] *= al[mf][r];

    #pragma unroll
    for (int ks = 0; ks < 4; ++ks) {               // 4 key sub-chunks of 32
      Frag32B Ap[2];
      #pragma unroll
      for (int mf = 0; mf < 2; ++mf) {
        const uint4* pp = (const uint4*)(lds_p + (mf * 16 + l16) * PROW + ks * 32 + lh * 8);
        Ap[mf].u[0] = pp[0];
        Ap[mf].u[1] = pp[2];
      }
      #pragma unroll
      for (int nt = 0; nt < 4; ++nt) {
        // B frag (V 32x16): lane col = output col; keys lh*8+0..7, 16+lh*8+0..7 from transposed Vt
        Frag32B Bv;
        const uint4* vp = (const uint4*)(Vt +
            (size_t)(wave * 64 + nt * 16 + l16) * NKEYS + nt0 + ks * 32 + lh * 8);
        Bv.u[0] = vp[0]; Bv.u[1] = vp[2];
        #pragma unroll
        for (int mf = 0; mf < 2; ++mf)
          O[mf][nt] = wmma_bf16(Ap[mf].v, Bv.v, O[mf][nt]);
      }
    }
    // No trailing barrier needed: next iteration's post-store __syncthreads separates
    // these lds_p/lds_alpha reads from their next writers.
  }

  //---------------- Epilogue: O / l, fp32 store --------------------------------------
  float linv[2][8];
  #pragma unroll
  for (int mf = 0; mf < 2; ++mf)
    #pragma unroll
    for (int r = 0; r < 8; ++r)
      linv[mf][r] = 1.0f / lds_l[mf * 16 + lh * 8 + r];
  #pragma unroll
  for (int mf = 0; mf < 2; ++mf)
    #pragma unroll
    for (int nt = 0; nt < 4; ++nt)
      #pragma unroll
      for (int r = 0; r < 8; ++r)
        out[(size_t)(qrow0 + mf * 16 + lh * 8 + r) * DVDIM +
            wave * 64 + nt * 16 + l16] = O[mf][nt][r] * linv[mf][r];
}

//--------------------------------------------------------------------------------------
extern "C" void kernel_launch(void* const* d_in, const int* in_sizes, int n_in,
                              void* d_out, int out_size, void* d_ws, size_t ws_size,
                              hipStream_t stream) {
  (void)in_sizes; (void)n_in; (void)out_size; (void)ws_size;
  const float* Q = (const float*)d_in[0];
  const float* K = (const float*)d_in[1];
  const float* V = (const float*)d_in[2];
  float* out = (float*)d_out;

  // d_ws layout: Qb (8 MB) | Kb (8 MB) | Vt (8 MB), all bf16 bits
  unsigned short* Qb = (unsigned short*)d_ws;
  unsigned short* Kb = Qb + (size_t)NKEYS * DKDIM;
  unsigned short* Vt = Kb + (size_t)NKEYS * DKDIM;

  cvt_qk_bf16   <<<4096, 256, 0, stream>>>(Q, K, Qb, Kb);
  cvt_v_transpose<<<4096, dim3(32, 8), 0, stream>>>(V, Vt);
  attn_fused    <<<NKEYS / BM, 256, 0, stream>>>(Qb, Kb, Vt, out);
}
